// LSTMCRF_4097398800924
// MI455X (gfx1250) — matrix operand order repaired and checked by reading, not compiled, not run on hardware
//
#include <hip/hip_runtime.h>
#include <hip/hip_bf16.h>
#include <stdint.h>

// ---------------------------------------------------------------------------
// LSTM+CRF for MI455X (gfx1250). wave32, WMMA bf16 16x16x32 (f32 accum),
// TDM tensor_load_to_lds for GEMM A-panel staging.
// B=64 T=512 V=50000 E=512 H=512 HD=256 L=2 K=16
// ---------------------------------------------------------------------------

typedef __attribute__((ext_vector_type(16))) __bf16 v16bf;
typedef __attribute__((ext_vector_type(8)))  __bf16 v8bf;
typedef __attribute__((ext_vector_type(4)))  __bf16 v4bf;
typedef __attribute__((ext_vector_type(8)))  float  v8f;
typedef __attribute__((ext_vector_type(4)))  float  v4f;
typedef __attribute__((ext_vector_type(4)))  unsigned u32x4;
typedef __attribute__((ext_vector_type(8)))  unsigned u32x8;

#define NB   64
#define NT   512
#define NE   512
#define NHD  256
#define NG   1024     // 4*HD
#define NK   16
#define TAG_START 13
#define TAG_STOP  14
#define NEGINF (-10000.0f)

__device__ __forceinline__ float sigf(float x) { return 1.0f / (1.0f + __expf(-x)); }

// ---------------------------------------------------------------------------
__global__ void k_cvt_bf16(const float* __restrict__ s, __bf16* __restrict__ d, int n) {
    int i = blockIdx.x * 256 + threadIdx.x;
    if (i < n) d[i] = (__bf16)s[i];
}

__global__ void k_bias(const float* __restrict__ a, const float* __restrict__ b,
                       float* __restrict__ d, int n) {
    int i = blockIdx.x * 256 + threadIdx.x;
    if (i < n) d[i] = a[i] + b[i];
}

// ---------------------------------------------------------------------------
// embedding gather -> bf16 activations (activations are only ever consumed as
// WMMA A operands, so store them in bf16 directly; halves the traffic)
__global__ void k_gather(const int* __restrict__ x, const float* __restrict__ tab,
                         __bf16* __restrict__ out) {
    int bt = blockIdx.x;
    int row = x[bt];
    v4f v = ((const v4f*)(tab + (size_t)row * NE))[threadIdx.x];
    v4bf o;
    for (int j = 0; j < 4; ++j) o[j] = (__bf16)v[j];
    ((v4bf*)(out + (size_t)bt * NE))[threadIdx.x] = o;
}

// ---------------------------------------------------------------------------
// Input projection GEMM: C[M,1024] = Abf[M,512] @ W[1024,512]^T + bias
// Block: 256 thr = 8 waves; block tile 16(M) x 128(N).
// A panel (16 x 512 bf16 = 16 KB) is DMA'd into LDS once by the Tensor Data
// Mover with LDS padding (pad 16B per 1024B row -> 1040B row stride, so the 16
// fragment rows hit distinct bank groups). K-loop is barrier-free pure WMMA.
#define AROW_STRIDE 520   // elements: 512 + 8 bf16 pad (TDM pad_amount = 4 DW)

__global__ void k_xproj(const __bf16* __restrict__ Abf, const __bf16* __restrict__ W,
                        const float* __restrict__ bias, float* __restrict__ C) {
    __shared__ __align__(1024) __bf16 aS[16 * AROW_STRIDE];
    const int tid  = threadIdx.x;
    const int wave = tid >> 5, lane = tid & 31, half = lane >> 4, nl = lane & 15;
    const int m0 = blockIdx.y * 16;
    const int n0 = blockIdx.x * 128 + wave * 16;

    if (wave == 0) {
        // Tensor DMA descriptor (ISA 8.3/8.4): 2D tile, data_size=2B,
        // tile 512x16, tensor stride 512, pad_enable, pad_interval=256DW(1024B),
        // pad_amount=4DW(16B).
        uint64_t ga = (uint64_t)(uintptr_t)(Abf + (size_t)m0 * 512);
        unsigned lds_off = (unsigned)(uintptr_t)(&aS[0]);  // LDS aperture: low 32b = LDS byte addr
        u32x4 g0 = { 1u,                                   // count=1 (valid D#)
                     lds_off,                              // lds_addr
                     (unsigned)ga,                         // global_addr[31:0]
                     (unsigned)((ga >> 32) & 0x1FFFFFFu) | 0x80000000u };  // addr[56:32] | type=2
        u32x8 g1 = { 0x07D10000u,   // data_size=1(2B) | pad_en | pad_interval=7 | pad_amount=3
                     0x02000000u,   // tensor_dim0 = 512 (bits 63:48)
                     0x00100000u,   // tensor_dim1 = 16  (bits 95:80)
                     0x02000000u,   // tile_dim0 = 512   (bits 127:112)
                     16u,           // tile_dim1 = 16
                     512u,          // tensor_dim0_stride = 512
                     0u, 0u };
        asm volatile("tensor_load_to_lds %0, %1" :: "s"(g0), "s"(g1) : "memory");
        __builtin_amdgcn_s_wait_tensorcnt(0);
    }
    __syncthreads();

    v8f acc = {};
    const __bf16* arow = aS + nl * AROW_STRIDE;
    for (int kk = 0; kk < 16; ++kk) {           // K = 512 in chunks of 32
        v8bf lo = *(const v8bf*)(arow + kk * 32 + half * 16);
        v8bf hi = *(const v8bf*)(arow + kk * 32 + half * 16 + 8);
        v16bf a;
        for (int e = 0; e < 8; ++e) { a[e] = lo[e]; a[8 + e] = hi[e]; }
        v16bf b = *(const v16bf*)(W + (size_t)(n0 + nl) * 512 + kk * 32 + half * 16);
        acc = __builtin_amdgcn_wmma_f32_16x16x32_bf16(false, a, false, b,
                                                      (short)0, acc, false, false);
    }
    for (int e = 0; e < 8; ++e) {
        int m = m0 + half * 8 + e;
        int n = n0 + nl;
        C[(size_t)m * NG + n] = acc[e] + bias[n];
    }
}

// ---------------------------------------------------------------------------
// Recurrent LSTM kernel. One block per direction (grid=2), 1024 thr = 32 waves.
// h lives in LDS as bf16; c + gate accumulators stay in registers. Each wave
// owns 2 (batch-tile, hd-tile) pairs; the 4 gate tiles (i,f,g,o) for a pair
// land in the same lanes -> LSTM cell entirely in registers.
__global__ __launch_bounds__(1024)
void k_lstm(const float* __restrict__ xprojF, const float* __restrict__ xprojB,
            const __bf16* __restrict__ WhhL,   // [2][1024][256]
            const float* __restrict__ h0L,     // [2][64][256]
            const float* __restrict__ c0L,     // [2][64][256]
            __bf16* __restrict__ outAct)       // [B,T,512] bf16; fwd 0..255, bwd 256..511
{
    const int dir = blockIdx.x;
    const float*  xproj = dir ? xprojB : xprojF;
    const __bf16* Whh   = WhhL + (size_t)dir * NG * NHD;
    const float*  h0d   = h0L + (size_t)dir * NB * NHD;
    const float*  c0d   = c0L + (size_t)dir * NB * NHD;

    __shared__ __align__(16) __bf16 hS[NB][264];   // 256 + 8 pad

    const int tid = threadIdx.x, wave = tid >> 5, lane = tid & 31;
    const int half = lane >> 4, nl = lane & 15;

    for (int i = tid; i < NB * NHD; i += 1024) {
        int b = i >> 8, hd = i & 255;
        hS[b][hd] = (__bf16)h0d[i];
    }

    int mt[2], ht[2];
    float c[2][8];
    v8f acc[2][4];
    for (int p = 0; p < 2; ++p) {
        int pr = wave * 2 + p;
        mt[p] = pr >> 4;
        ht[p] = pr & 15;
        for (int e = 0; e < 8; ++e) {
            int b = mt[p] * 16 + half * 8 + e;
            c[p][e] = c0d[b * NHD + ht[p] * 16 + nl];
        }
    }
    __syncthreads();

    for (int t = 0; t < NT; ++t) {
        int tt = dir ? (NT - 1 - t) : t;
        for (int p = 0; p < 2; ++p)
            for (int g = 0; g < 4; ++g)
                for (int e = 0; e < 8; ++e) {
                    int b = mt[p] * 16 + half * 8 + e;
                    acc[p][g][e] = xproj[((size_t)b * NT + tt) * NG + g * NHD + ht[p] * 16 + nl];
                }
        if (t + 1 < NT) {
            int tn = dir ? (NT - 2 - t) : (t + 1);
            __builtin_prefetch(&xproj[((size_t)(mt[0] * 16 + half * 8) * NT + tn) * NG +
                                      ht[0] * 16 + nl], 0, 0);
        }
        for (int kk = 0; kk < 8; ++kk) {        // K = 256 in chunks of 32
            for (int p = 0; p < 2; ++p) {
                int row = mt[p] * 16 + nl;
                v8bf lo = *(const v8bf*)&hS[row][kk * 32 + half * 16];
                v8bf hi = *(const v8bf*)&hS[row][kk * 32 + half * 16 + 8];
                v16bf a;
                for (int e = 0; e < 8; ++e) { a[e] = lo[e]; a[8 + e] = hi[e]; }
                for (int g = 0; g < 4; ++g) {
                    const __bf16* wp = Whh + (size_t)(g * NHD + ht[p] * 16 + nl) * NHD +
                                       kk * 32 + half * 16;
                    v16bf bfr = *(const v16bf*)wp;
                    acc[p][g] = __builtin_amdgcn_wmma_f32_16x16x32_bf16(
                        false, a, false, bfr, (short)0, acc[p][g], false, false);
                }
            }
        }
        __syncthreads();   // all hS reads done before overwrite
        for (int p = 0; p < 2; ++p) {
            for (int e = 0; e < 8; ++e) {
                float gi = acc[p][0][e], gf = acc[p][1][e];
                float gg = acc[p][2][e], go = acc[p][3][e];
                float cc = sigf(gf) * c[p][e] + sigf(gi) * tanhf(gg);
                c[p][e] = cc;
                float hh = sigf(go) * tanhf(cc);
                int b  = mt[p] * 16 + half * 8 + e;
                int hd = ht[p] * 16 + nl;
                __bf16 hb = (__bf16)hh;
                hS[b][hd] = hb;
                outAct[((size_t)b * NT + tt) * 512 + dir * NHD + hd] = hb;
            }
        }
        __syncthreads();
    }
}

// ---------------------------------------------------------------------------
// tag_scores = act[M,512](bf16) @ w_out[16,512]^T + b_out  (N=16 -> one tile)
__global__ void k_tag(const __bf16* __restrict__ Abf, const __bf16* __restrict__ Wout,
                      const float* __restrict__ bout, float* __restrict__ out) {
    const int tid = threadIdx.x, wave = tid >> 5, lane = tid & 31;
    const int half = lane >> 4, nl = lane & 15;
    const int m0 = (blockIdx.x * 8 + wave) * 16;
    v8f acc = {};
    const __bf16* arow = Abf + (size_t)(m0 + nl) * 512;
    for (int kk = 0; kk < 16; ++kk) {
        v8bf lo = *(const v8bf*)(arow + kk * 32 + half * 16);
        v8bf hi = *(const v8bf*)(arow + kk * 32 + half * 16 + 8);
        v16bf a;
        for (int e = 0; e < 8; ++e) { a[e] = lo[e]; a[8 + e] = hi[e]; }
        v16bf b = *(const v16bf*)(Wout + (size_t)nl * 512 + kk * 32 + half * 16);
        acc = __builtin_amdgcn_wmma_f32_16x16x32_bf16(false, a, false, b,
                                                      (short)0, acc, false, false);
    }
    for (int e = 0; e < 8; ++e) {
        int m = m0 + half * 8 + e;
        out[(size_t)m * NK + nl] = acc[e] + bout[nl];
    }
}

// ---------------------------------------------------------------------------
// CRF forward pass, faithful to reference (cross-batch exp-sum), deterministic.
__global__ __launch_bounds__(1024)
void k_crf(const float* __restrict__ feat,   // [B,T,16]
           const float* __restrict__ trans,  // [16,16]
           float* __restrict__ fscore)       // [B]
{
    __shared__ float alpha[NB][NK];
    __shared__ float tr[NK][NK];
    __shared__ float part[NB][NK];
    __shared__ float sv[NK];
    __shared__ float mrow[NB];
    __shared__ float pb[NB];
    __shared__ float tot;

    const int tid = threadIdx.x;
    const int b = tid >> 4, kn = tid & 15;
    if (tid < NK * NK) ((float*)tr)[tid] = trans[tid];
    alpha[b][kn] = (kn == TAG_START) ? 0.0f : NEGINF;
    __syncthreads();

    for (int t = 0; t < NT; ++t) {
        float f = feat[((size_t)b * NT + t) * NK + kn];
        float v[NK];
        float m = -3.4e38f;
        for (int kp = 0; kp < NK; ++kp) {
            v[kp] = alpha[b][kp] + f + tr[kn][kp];
            m = fmaxf(m, v[kp]);
        }
        float p = 0.0f;
        for (int kp = 0; kp < NK; ++kp) p += __expf(v[kp] - m);
        part[b][kn] = p;
        __syncthreads();
        if (tid < NK) {
            float s = 0.0f;
            for (int bb = 0; bb < NB; ++bb) s += part[bb][tid];
            sv[tid] = __logf(s);
        }
        __syncthreads();
        float na = m + sv[kn];
        alpha[b][kn] = na;
        __syncthreads();
    }
    part[b][kn] = alpha[b][kn] + tr[TAG_STOP][kn];
    __syncthreads();
    if (tid < NB) {
        float m = -3.4e38f;
        for (int k = 0; k < NK; ++k) m = fmaxf(m, part[tid][k]);
        mrow[tid] = m;
        float p = 0.0f;
        for (int k = 0; k < NK; ++k) p += __expf(part[tid][k] - m);
        pb[tid] = p;
    }
    __syncthreads();
    if (tid == 0) {
        float s = 0.0f;
        for (int bb = 0; bb < NB; ++bb) s += pb[bb];
        tot = __logf(s);
    }
    __syncthreads();
    if (tid < NB) fscore[tid] = mrow[tid] + tot;
}

// ---------------------------------------------------------------------------
extern "C" void kernel_launch(void* const* d_in, const int* in_sizes, int n_in,
                              void* d_out, int out_size, void* d_ws, size_t ws_size,
                              hipStream_t stream) {
    const int*   x         = (const int*)d_in[0];
    const float* emb_table = (const float*)d_in[2];
    const float* w_ih      = (const float*)d_in[3];
    const float* w_hh      = (const float*)d_in[4];
    const float* b_ih      = (const float*)d_in[5];
    const float* b_hh      = (const float*)d_in[6];
    const float* h0        = (const float*)d_in[7];
    const float* c0        = (const float*)d_in[8];
    const float* w_out     = (const float*)d_in[9];
    const float* b_out     = (const float*)d_in[10];
    const float* trans     = (const float*)d_in[11];
    float* out = (float*)d_out;

    char* ws = (char*)d_ws;
    size_t off = 0;
    __bf16* actbf0 = (__bf16*)(ws + off); off += (size_t)NB * NT * 512 * 2;     // 32 MB
    __bf16* actbf1 = (__bf16*)(ws + off); off += (size_t)NB * NT * 512 * 2;     // 32 MB
    float*  xpf    = (float*)(ws + off);  off += (size_t)NB * NT * NG * 4;      // 128 MB
    float*  xpb    = (float*)(ws + off);  off += (size_t)NB * NT * NG * 4;      // 128 MB
    __bf16* wihb   = (__bf16*)(ws + off); off += (size_t)2 * 2 * NG * NE * 2;   // 4 MB
    __bf16* whhb   = (__bf16*)(ws + off); off += (size_t)2 * 2 * NG * NHD * 2;  // 2 MB
    __bf16* woutb  = (__bf16*)(ws + off); off += (size_t)NK * 512 * 2;
    float*  bias   = (float*)(ws + off);  off += (size_t)2 * 2 * NG * 4;

    int n;
    n = 2 * 2 * NG * NE;  k_cvt_bf16<<<(n + 255) / 256, 256, 0, stream>>>(w_ih, wihb, n);
    n = 2 * 2 * NG * NHD; k_cvt_bf16<<<(n + 255) / 256, 256, 0, stream>>>(w_hh, whhb, n);
    n = NK * 512;         k_cvt_bf16<<<(n + 255) / 256, 256, 0, stream>>>(w_out, woutb, n);
    n = 2 * 2 * NG;       k_bias<<<(n + 255) / 256, 256, 0, stream>>>(b_ih, b_hh, bias, n);

    k_gather<<<NB * NT, 128, 0, stream>>>(x, emb_table, actbf0);

    const int M = NB * NT;                 // 32768 rows
    for (int l = 0; l < 2; ++l) {
        const __bf16* in = l ? actbf1 : actbf0;
        __bf16*       oa = l ? actbf0 : actbf1;
        k_xproj<<<dim3(8, M / 16), 256, 0, stream>>>(
            in, wihb + (size_t)(l * 2 + 0) * NG * NE, bias + (l * 2 + 0) * NG, xpf);
        k_xproj<<<dim3(8, M / 16), 256, 0, stream>>>(
            in, wihb + (size_t)(l * 2 + 1) * NG * NE, bias + (l * 2 + 1) * NG, xpb);
        k_lstm<<<2, 1024, 0, stream>>>(
            xpf, xpb, whhb + (size_t)l * 2 * NG * NHD,
            h0 + (size_t)2 * l * NB * NHD, c0 + (size_t)2 * l * NB * NHD, oa);
    }

    k_tag<<<M / (8 * 16), 256, 0, stream>>>(actbf0, woutb, b_out, out);
    k_crf<<<1, 1024, 0, stream>>>(out, trans, out + (size_t)NB * NT * NK);
}
